// KeyValueMemory_12850542150220
// MI455X (gfx1250) — matrix operand order repaired and checked by reading, not compiled, not run on hardware
//
#include <hip/hip_runtime.h>
#include <hip/hip_bf16.h>

// ---------------------------------------------------------------------------
// Problem constants (from reference)
// ---------------------------------------------------------------------------
#define B_SZ   1024
#define N_SZ   65536
#define K_SZ   1024
#define DOUT_  1024
#define THRESH 0.1f
#define NVAR   1e-4f

// Tiling for the fused WMMA kernel
#define BM     64          // rows of B (queries) per block
#define BN     64          // keys per N-chunk
#define NITER  8           // N-chunks per block  -> 512 keys per block
#define KSTEP  32          // K per WMMA (bf16)
#define NK     (K_SZ / KSTEP)
#define LDSROW 80          // bytes per 32-bf16 LDS row (64B data + 16B pad)

typedef __attribute__((ext_vector_type(16))) __bf16 v16bf;
typedef __attribute__((ext_vector_type(8)))  float  v8f;

union FragBF {
    v16bf v;
    uint4 q[2];
};

// ---------------------------------------------------------------------------
// Helpers: packed fp32 -> bf16 (RNE) and hi/lo split
// ---------------------------------------------------------------------------
__device__ __forceinline__ unsigned pk2bf(float a, float b) {
    union { __hip_bfloat162 h; unsigned u; } v;
    v.h = __float22bfloat162_rn(make_float2(a, b));   // -> v_cvt_pk_bf16_f32
    return v.u;
}
__device__ __forceinline__ float asf(unsigned u) {
    union { float f; unsigned u; } v; v.u = u; return v.f;
}
__device__ __forceinline__ void cvt_split(float4 f, uint2& hi, uint2& lo) {
    hi.x = pk2bf(f.x, f.y);
    hi.y = pk2bf(f.z, f.w);
    const float h0 = asf(hi.x << 16);
    const float h1 = asf(hi.x & 0xFFFF0000u);
    const float h2 = asf(hi.y << 16);
    const float h3 = asf(hi.y & 0xFFFF0000u);
    lo.x = pk2bf(f.x - h0, f.y - h1);
    lo.y = pk2bf(f.z - h2, f.w - h3);
}
// monotonic float -> u32 (order-preserving, incl. negatives)
__device__ __forceinline__ unsigned orderedF32(float f) {
    union { float f; unsigned u; } v; v.f = f;
    return (v.u & 0x80000000u) ? ~v.u : (v.u | 0x80000000u);
}

// ---------------------------------------------------------------------------
// Kernel 1: row squared-norms (one wave32 per row, K_SZ = 1024)
// ---------------------------------------------------------------------------
__global__ __launch_bounds__(256) void rownorm_kernel(
    const float* __restrict__ src, float* __restrict__ dst) {
    const int lane = threadIdx.x & 31;
    const int row  = blockIdx.x * 8 + (threadIdx.x >> 5);
    const float* p = src + (size_t)row * K_SZ;
    float s = 0.f;
#pragma unroll
    for (int i = 0; i < 8; ++i) {
        float4 v = *(const float4*)(p + lane * 4 + i * 128);
        s += v.x * v.x + v.y * v.y + v.z * v.z + v.w * v.w;
    }
#pragma unroll
    for (int off = 16; off; off >>= 1) s += __shfl_xor(s, off, 32);
    if (lane == 0) dst[row] = s;
}

// ---------------------------------------------------------------------------
// Kernel 2: init per-row argmax accumulators (must run every call)
// ---------------------------------------------------------------------------
__global__ void init_kernel(unsigned long long* __restrict__ packed,
                            unsigned* __restrict__ flags) {
    int i = blockIdx.x * 256 + threadIdx.x;
    if (i < B_SZ) { packed[i] = 0ull; flags[i] = 0u; }
}

// ---------------------------------------------------------------------------
// Kernel 3: fused dual-GEMM (bf16x3 split WMMA) + gate + score + argmax
//   grid = (B_SZ/BM, N_SZ/(BN*NITER)) ; block = 256 threads = 8 waves
//   wave grid: 4 (M) x 2 (N-pair); each wave: 16 rows x 32 cols, 2 GEMMs
//   Double-buffered LDS: one barrier per K-step; global prefetch overlaps WMMA
// ---------------------------------------------------------------------------
__global__ __launch_bounds__(256) void fused_gemm_argmax_kernel(
    const float* __restrict__ X,  const float* __restrict__ Y,
    const float* __restrict__ xk, const float* __restrict__ yk,
    const float* __restrict__ noise,
    const float* __restrict__ y_norms, const float* __restrict__ q_norms,
    unsigned long long* __restrict__ packed, unsigned* __restrict__ flags) {

    __shared__ __align__(16) unsigned char sAxh[2][BM * LDSROW];
    __shared__ __align__(16) unsigned char sAxl[2][BM * LDSROW];
    __shared__ __align__(16) unsigned char sAyh[2][BM * LDSROW];
    __shared__ __align__(16) unsigned char sAyl[2][BM * LDSROW];
    __shared__ __align__(16) unsigned char sBxh[2][BN * LDSROW];
    __shared__ __align__(16) unsigned char sBxl[2][BN * LDSROW];
    __shared__ __align__(16) unsigned char sByh[2][BN * LDSROW];
    __shared__ __align__(16) unsigned char sByl[2][BN * LDSROW];
    __shared__ unsigned long long sRed[BM];
    __shared__ unsigned           sGate[BM];

    const int tid   = threadIdx.x;
    const int lane  = tid & 31;
    const int wave  = tid >> 5;
    const int wm    = wave & 3;        // 0..3  (M sub-tile, 16 rows each)
    const int wn    = wave >> 2;       // 0..1  (N pair, 32 cols each)
    const int lhalf = lane >> 4;       // 0..1
    const int l15   = lane & 15;
    const int bm0   = blockIdx.x * BM;
    const int nb0   = blockIdx.y * (BN * NITER);

    if (tid < BM) { sRed[tid] = 0ull; sGate[tid] = 0u; }

    // staging job coordinates: two jobs per thread, rows r0 and r0+32
    const int r0 = tid >> 3;                 // 0..31
    const int r1 = r0 + 32;                  // 32..63
    const int c4 = (tid & 7) * 4;            // 0,4,...,28
    const int sO0 = r0 * LDSROW + c4 * 2;    // LDS byte offset job0
    const int sO1 = r1 * LDSROW + c4 * 2;    // LDS byte offset job1

    // hoisted per-row query norms (fixed across the whole kernel)
    float qn[8];
#pragma unroll
    for (int j = 0; j < 8; ++j)
        qn[j] = q_norms[bm0 + wm * 16 + j + lhalf * 8];

    // running per-row best across all N handled by this block
    unsigned bestU[8], bestN[8], gAny = 0u;
#pragma unroll
    for (int j = 0; j < 8; ++j) { bestU[j] = 0u; bestN[j] = 0u; }

    // fragment LDS coordinates (constant per thread)
    const int arow  = wm * 16 + l15;
    const int aoff  = arow * LDSROW + lhalf * 16;   // lanes>=16: K 8..15 / 24..31
    const int boff0 = (wn * 32 + l15) * LDSROW + lhalf * 16;   // N-tile 0
    const int boff1 = boff0 + 16 * LDSROW;                     // N-tile 1

    // global prefetch registers (8 float4)
    float4 gX0, gX1, gY0, gY1, gKx0, gKx1, gKy0, gKy1;

#define LOADG(k0_, bn0_)                                                     \
    do {                                                                     \
        const size_t oA0 = (size_t)(bm0 + r0) * K_SZ + (k0_) + c4;           \
        const size_t oA1 = (size_t)(bm0 + r1) * K_SZ + (k0_) + c4;           \
        const size_t oB0 = (size_t)((bn0_) + r0) * K_SZ + (k0_) + c4;        \
        const size_t oB1 = (size_t)((bn0_) + r1) * K_SZ + (k0_) + c4;        \
        gX0  = *(const float4*)(X  + oA0);  gX1  = *(const float4*)(X  + oA1); \
        gY0  = *(const float4*)(Y  + oA0);  gY1  = *(const float4*)(Y  + oA1); \
        gKx0 = *(const float4*)(xk + oB0);  gKx1 = *(const float4*)(xk + oB1); \
        gKy0 = *(const float4*)(yk + oB0);  gKy1 = *(const float4*)(yk + oB1); \
    } while (0)

#define STAGE(p_)                                                            \
    do {                                                                     \
        uint2 h, l;                                                          \
        cvt_split(gX0,  h, l); *(uint2*)(&sAxh[p_][sO0]) = h; *(uint2*)(&sAxl[p_][sO0]) = l; \
        cvt_split(gX1,  h, l); *(uint2*)(&sAxh[p_][sO1]) = h; *(uint2*)(&sAxl[p_][sO1]) = l; \
        cvt_split(gY0,  h, l); *(uint2*)(&sAyh[p_][sO0]) = h; *(uint2*)(&sAyl[p_][sO0]) = l; \
        cvt_split(gY1,  h, l); *(uint2*)(&sAyh[p_][sO1]) = h; *(uint2*)(&sAyl[p_][sO1]) = l; \
        cvt_split(gKx0, h, l); *(uint2*)(&sBxh[p_][sO0]) = h; *(uint2*)(&sBxl[p_][sO0]) = l; \
        cvt_split(gKx1, h, l); *(uint2*)(&sBxh[p_][sO1]) = h; *(uint2*)(&sBxl[p_][sO1]) = l; \
        cvt_split(gKy0, h, l); *(uint2*)(&sByh[p_][sO0]) = h; *(uint2*)(&sByl[p_][sO0]) = l; \
        cvt_split(gKy1, h, l); *(uint2*)(&sByh[p_][sO1]) = h; *(uint2*)(&sByl[p_][sO1]) = l; \
    } while (0)

    for (int ni = 0; ni < NITER; ++ni) {
        const int bn0 = nb0 + ni * BN;
        v8f accx0 = {0.f,0.f,0.f,0.f,0.f,0.f,0.f,0.f};
        v8f accx1 = accx0, accy0 = accx0, accy1 = accx0;

        LOADG(0, bn0);                         // prologue prefetch

        for (int kt = 0; kt < NK; ++kt) {
            const int p = kt & 1;
            STAGE(p);                          // convert + store current step
            if (kt + 1 < NK)                   // prefetch next step; overlaps
                LOADG((kt + 1) * KSTEP, bn0);  //   barrier + WMMA below
            __syncthreads();                   // single barrier per K-step

            // ---- fragment loads per ISA 16-bit A / B layouts (wave32) ----
            FragBF axh, axl, ayh, ayl;
            FragBF bxh0, bxl0, byh0, byl0, bxh1, bxl1, byh1, byl1;
            axh.q[0]  = *(const uint4*)(&sAxh[p][aoff]);
            axh.q[1]  = *(const uint4*)(&sAxh[p][aoff + 32]);
            axl.q[0]  = *(const uint4*)(&sAxl[p][aoff]);
            axl.q[1]  = *(const uint4*)(&sAxl[p][aoff + 32]);
            ayh.q[0]  = *(const uint4*)(&sAyh[p][aoff]);
            ayh.q[1]  = *(const uint4*)(&sAyh[p][aoff + 32]);
            ayl.q[0]  = *(const uint4*)(&sAyl[p][aoff]);
            ayl.q[1]  = *(const uint4*)(&sAyl[p][aoff + 32]);
            bxh0.q[0] = *(const uint4*)(&sBxh[p][boff0]);
            bxh0.q[1] = *(const uint4*)(&sBxh[p][boff0 + 32]);
            bxl0.q[0] = *(const uint4*)(&sBxl[p][boff0]);
            bxl0.q[1] = *(const uint4*)(&sBxl[p][boff0 + 32]);
            byh0.q[0] = *(const uint4*)(&sByh[p][boff0]);
            byh0.q[1] = *(const uint4*)(&sByh[p][boff0 + 32]);
            byl0.q[0] = *(const uint4*)(&sByl[p][boff0]);
            byl0.q[1] = *(const uint4*)(&sByl[p][boff0 + 32]);
            bxh1.q[0] = *(const uint4*)(&sBxh[p][boff1]);
            bxh1.q[1] = *(const uint4*)(&sBxh[p][boff1 + 32]);
            bxl1.q[0] = *(const uint4*)(&sBxl[p][boff1]);
            bxl1.q[1] = *(const uint4*)(&sBxl[p][boff1 + 32]);
            byh1.q[0] = *(const uint4*)(&sByh[p][boff1]);
            byh1.q[1] = *(const uint4*)(&sByh[p][boff1 + 32]);
            byl1.q[0] = *(const uint4*)(&sByl[p][boff1]);
            byl1.q[1] = *(const uint4*)(&sByl[p][boff1 + 32]);

            // ---- bf16x3: hi*hi + hi*lo + lo*hi  (fp32-class accuracy) ----
            accx0 = __builtin_amdgcn_wmma_f32_16x16x32_bf16(false, axh.v, false, bxh0.v, (short)0, accx0, false, false);
            accx0 = __builtin_amdgcn_wmma_f32_16x16x32_bf16(false, axh.v, false, bxl0.v, (short)0, accx0, false, false);
            accx0 = __builtin_amdgcn_wmma_f32_16x16x32_bf16(false, axl.v, false, bxh0.v, (short)0, accx0, false, false);
            accx1 = __builtin_amdgcn_wmma_f32_16x16x32_bf16(false, axh.v, false, bxh1.v, (short)0, accx1, false, false);
            accx1 = __builtin_amdgcn_wmma_f32_16x16x32_bf16(false, axh.v, false, bxl1.v, (short)0, accx1, false, false);
            accx1 = __builtin_amdgcn_wmma_f32_16x16x32_bf16(false, axl.v, false, bxh1.v, (short)0, accx1, false, false);
            accy0 = __builtin_amdgcn_wmma_f32_16x16x32_bf16(false, ayh.v, false, byh0.v, (short)0, accy0, false, false);
            accy0 = __builtin_amdgcn_wmma_f32_16x16x32_bf16(false, ayh.v, false, byl0.v, (short)0, accy0, false, false);
            accy0 = __builtin_amdgcn_wmma_f32_16x16x32_bf16(false, ayl.v, false, byh0.v, (short)0, accy0, false, false);
            accy1 = __builtin_amdgcn_wmma_f32_16x16x32_bf16(false, ayh.v, false, byh1.v, (short)0, accy1, false, false);
            accy1 = __builtin_amdgcn_wmma_f32_16x16x32_bf16(false, ayh.v, false, byl1.v, (short)0, accy1, false, false);
            accy1 = __builtin_amdgcn_wmma_f32_16x16x32_bf16(false, ayl.v, false, byh1.v, (short)0, accy1, false, false);
        }

        // ---- epilogue: gate + score + running per-row argmax ----
        // C/D layout: VGPR j, lanes 0-15 -> M=j ; lanes 16-31 -> M=j+8 ; N=lane&15
        const int n0 = bn0 + wn * 32 + l15;
        const int n1 = n0 + 16;
        const float yn0 = y_norms[n0];
        const float yn1 = y_norms[n1];
#pragma unroll
        for (int j = 0; j < 8; ++j) {
            const int m = bm0 + wm * 16 + j + lhalf * 8;
            // N-tile 0
            {
                const float kqx = accx0[j], kqy = accy0[j];
                const float dy = yn0 + qn[j] - 2.0f * kqy;
                const bool  gate = (dy < THRESH) && (kqx > 0.0f);
                const float score = (gate ? (kqx + kqy) : 0.0f)
                                  + noise[(size_t)m * N_SZ + n0] * NVAR;
                const unsigned u = orderedF32(score);
                if (u > bestU[j]) { bestU[j] = u; bestN[j] = (unsigned)n0; }
                gAny |= gate ? (1u << j) : 0u;
            }
            // N-tile 1
            {
                const float kqx = accx1[j], kqy = accy1[j];
                const float dy = yn1 + qn[j] - 2.0f * kqy;
                const bool  gate = (dy < THRESH) && (kqx > 0.0f);
                const float score = (gate ? (kqx + kqy) : 0.0f)
                                  + noise[(size_t)m * N_SZ + n1] * NVAR;
                const unsigned u = orderedF32(score);
                if (u > bestU[j]) { bestU[j] = u; bestN[j] = (unsigned)n1; }
                gAny |= gate ? (1u << j) : 0u;
            }
        }
    }

    // ---- cross-lane (16-lane group) reduce, then LDS, then global atomics ----
#pragma unroll
    for (int j = 0; j < 8; ++j) {
        unsigned u = bestU[j], n = bestN[j], g = (gAny >> j) & 1u;
#pragma unroll
        for (int off = 1; off < 16; off <<= 1) {
            unsigned uo = __shfl_xor(u, off, 32);
            unsigned no = __shfl_xor(n, off, 32);
            unsigned go = __shfl_xor(g, off, 32);
            if (uo > u || (uo == u && no > n)) { u = uo; n = no; }
            g |= go;
        }
        if (l15 == 0) {
            const int mLoc = wm * 16 + j + lhalf * 8;
            atomicMax(&sRed[mLoc], ((unsigned long long)u << 32) | (unsigned long long)n);
            if (g) atomicOr(&sGate[mLoc], 1u);
        }
    }
    __syncthreads();
    if (tid < BM) {
        atomicMax(&packed[bm0 + tid], sRed[tid]);
        if (sGate[tid]) atomicOr(&flags[bm0 + tid], 1u);
    }
#undef LOADG
#undef STAGE
}

// ---------------------------------------------------------------------------
// Kernel 4: gather x_hat[b] = any_gate * values[idx]
// ---------------------------------------------------------------------------
__global__ __launch_bounds__(256) void gather_kernel(
    const float* __restrict__ values,
    const unsigned long long* __restrict__ packed,
    const unsigned* __restrict__ flags,
    float* __restrict__ out) {
    const int b = blockIdx.x;
    const unsigned long long p = packed[b];
    const unsigned idx = (unsigned)(p & 0xFFFFFFFFull);
    const float s = flags[b] ? 1.0f : 0.0f;
    float4 v = *(const float4*)(values + (size_t)idx * DOUT_ + threadIdx.x * 4);
    float4 r; r.x = v.x * s; r.y = v.y * s; r.z = v.z * s; r.w = v.w * s;
    *(float4*)(out + (size_t)b * DOUT_ + threadIdx.x * 4) = r;
}

// ---------------------------------------------------------------------------
// Host launcher
// ---------------------------------------------------------------------------
extern "C" void kernel_launch(void* const* d_in, const int* in_sizes, int n_in,
                              void* d_out, int out_size, void* d_ws, size_t ws_size,
                              hipStream_t stream) {
    (void)in_sizes; (void)n_in; (void)out_size; (void)ws_size;
    const float* X      = (const float*)d_in[0];  // (B, 1024)
    const float* Y      = (const float*)d_in[1];  // (B, 1024)
    const float* xk     = (const float*)d_in[2];  // (N, 1024)
    const float* yk     = (const float*)d_in[3];  // (N, 1024)
    const float* values = (const float*)d_in[4];  // (N, 1024)
    const float* noise  = (const float*)d_in[5];  // (B, N)
    float*       out    = (float*)d_out;          // (B, 1024)

    // workspace layout
    unsigned long long* packed  = (unsigned long long*)d_ws;                 // 8 KB
    unsigned*           flags   = (unsigned*)((char*)d_ws + 8192);           // 4 KB
    float*              q_norms = (float*)((char*)d_ws + 12288);             // 4 KB
    float*              y_norms = (float*)((char*)d_ws + 16384);             // 256 KB

    // 1) norms
    rownorm_kernel<<<N_SZ / 8, 256, 0, stream>>>(yk, y_norms);
    rownorm_kernel<<<B_SZ / 8, 256, 0, stream>>>(Y,  q_norms);
    // 2) init argmax accumulators (every call)
    init_kernel<<<(B_SZ + 255) / 256, 256, 0, stream>>>(packed, flags);
    // 3) fused dual-GEMM + argmax ; x = B-chunks (fast) so N-chunk key slices
    //    are shared through L2 by the 16 co-scheduled row-blocks.
    dim3 grid(B_SZ / BM, N_SZ / (BN * NITER));
    fused_gemm_argmax_kernel<<<grid, 256, 0, stream>>>(
        X, Y, xk, yk, noise, y_norms, q_norms, packed, flags);
    // 4) gather
    gather_kernel<<<B_SZ, 256, 0, stream>>>(values, packed, flags, out);
}
